// DIENLayer_68418829025942
// MI455X (gfx1250) — matrix operand order repaired and checked by reading, not compiled, not run on hardware
//
#include <hip/hip_runtime.h>
#include <hip/hip_bf16.h>

typedef __attribute__((ext_vector_type(16))) __bf16 v16bf;
typedef __attribute__((ext_vector_type(8)))  float  v8f;

__device__ __forceinline__ float sigmoidf(float x) { return 1.f / (1.f + __expf(-x)); }

__device__ __forceinline__ v8f wmma_bf16(v16bf a, v16bf b, v8f c) {
  // (neg_a, A, neg_b, B, c_mod, C, reuse_a, reuse_b)
  return __builtin_amdgcn_wmma_f32_16x16x32_bf16(false, a, false, b, (short)0, c, false, false);
}

union FragU { v16bf v; uint4 u[2]; };

// A fragment: 16 rows x 32 K, p points at [row0][k0] of row-major bf16 buffer, ld = row stride (elems)
// ISA 7.12.2: lane<16 -> M=lane, K in {0..7,16..23}; lane>=16 -> M=lane-16, K in {8..15,24..31}
__device__ __forceinline__ v16bf load_A(const __bf16* p, int ld) {
  int lane = threadIdx.x & 31;
  const __bf16* r = p + (lane & 15) * ld + ((lane >> 4) << 3);
  FragU f;
  f.u[0] = *(const uint4*)(r);
  f.u[1] = *(const uint4*)(r + 16);
  return f.v;
}

// B fragment: 32 K x 16 N from TRANSPOSED buffer wT[N][K], ld = row stride (elems)
// lane<16 -> N=lane, K in {0..7,16..23}; lane>=16 -> N=lane-16, K in {8..15,24..31}
__device__ __forceinline__ v16bf load_B(const __bf16* p, int ld) {
  int lane = threadIdx.x & 31;
  const __bf16* r = p + (lane & 15) * ld + ((lane >> 4) << 3);
  FragU f;
  f.u[0] = *(const uint4*)(r);
  f.u[1] = *(const uint4*)(r + 16);
  return f.v;
}

// ---------------- seq_len ----------------
__global__ void k_seqlen(const int* __restrict__ mask, int* __restrict__ sl, int Bn, int Tn) {
  int b = blockIdx.x * blockDim.x + threadIdx.x;
  if (b >= Bn) return;
  int s = Tn;
  for (int t = 0; t < Tn; ++t) {
    if (mask[(size_t)b * Tn + t] == 0) { s = t; break; }
  }
  sl[b] = s;
}

// ---------------- qp = prelu(q @ wq + bq) ----------------
__global__ void k_qp(const float* __restrict__ item_eb, const float* __restrict__ wq,
                     const float* __restrict__ bq, const float* __restrict__ pal,
                     float* __restrict__ qp, int Bn) {
  int i = blockIdx.x * blockDim.x + threadIdx.x;
  if (i >= Bn * 128) return;
  int b = i >> 7, n = i & 127;
  const float* q = item_eb + (size_t)b * 128;
  float acc = bq[n];
  for (int k = 0; k < 128; ++k) acc += q[k] * wq[k * 128 + n];
  qp[i] = fmaxf(acc, 0.f) + pal[n] * fminf(acc, 0.f);
}

// ---------------- GRU (both passes), 16 batch rows per workgroup, 8 waves ----------------
// LDS layout (bf16 elems unless noted):
//   gkT[256][264]  ckT[128][264]  xb[16][392] (cols 0..127=x, 128..255=h, 256..383=r*h)
//   f32: hbuf[16][128] ubuf[16][128] gbs[256] cbs[128] av[16] ; int sl[16]
template<bool ATT>
__global__ __launch_bounds__(256) void k_gru(
    const float* __restrict__ xs, const float* __restrict__ gk, const float* __restrict__ gb,
    const float* __restrict__ ck, const float* __restrict__ cb, const int* __restrict__ seqlen,
    const float* __restrict__ att, float* __restrict__ outs, float* __restrict__ hfin, int Tn) {
  extern __shared__ char smem[];
  __bf16* gkT = (__bf16*)smem;            // 256*264
  __bf16* ckT = gkT + 256 * 264;          // 128*264
  __bf16* xb  = ckT + 128 * 264;          // 16*392
  float*  hbuf = (float*)(xb + 16 * 392); // 16*128
  float*  ubuf = hbuf + 16 * 128;         // 16*128
  float*  gbs  = ubuf + 16 * 128;         // 256
  float*  cbs  = gbs + 256;               // 128
  float*  av   = cbs + 128;               // 16
  int*    sl   = (int*)(av + 16);         // 16

  const int tid = threadIdx.x, wv = tid >> 5, lane = tid & 31;
  const int b0 = blockIdx.x * 16;
  const int cn = lane & 15, mo = (lane >> 4) ? 8 : 0;

  // Load weights once: transpose to [N][K] bf16 (reused across all 200 steps)
  for (int i = tid; i < 256 * 256; i += 256) {
    int k = i >> 8, n = i & 255;
    gkT[n * 264 + k] = (__bf16)gk[i];
  }
  for (int i = tid; i < 256 * 128; i += 256) {
    int k = i >> 7, n = i & 127;
    ckT[n * 264 + k] = (__bf16)ck[i];
  }
  gbs[tid] = gb[tid];
  if (tid < 128) cbs[tid] = cb[tid];
  if (tid < 16) sl[tid] = seqlen[b0 + tid];
  for (int i = tid; i < 16 * 128; i += 256) {
    int r = i >> 7, c = i & 127;
    hbuf[i] = 0.f;
    xb[r * 392 + 128 + c] = (__bf16)0.f;
  }
  __syncthreads();

  const __bf16* bp0 = gkT + (wv * 2 + 0) * 16 * 264;
  const __bf16* bp1 = gkT + (wv * 2 + 1) * 16 * 264;
  const __bf16* cp  = ckT + wv * 16 * 264;

  for (int t = 0; t < Tn; ++t) {
    // load x tile [16][128] -> bf16 LDS; prefetch next step
    for (int i = tid; i < 16 * 128; i += 256) {
      int r = i >> 7, c = i & 127;
      const float* px = &xs[((size_t)(b0 + r) * Tn + t) * 128 + c];
      xb[r * 392 + c] = (__bf16)(*px);
      if (t + 1 < Tn) __builtin_prefetch(px + 128, 0, 0);
    }
    if (ATT && tid < 16) av[tid] = att[(size_t)(b0 + tid) * Tn + t];
    __syncthreads();

    // z = sigmoid([x,h] @ gk + gb): wave wv owns N-tiles 2wv, 2wv+1.
    // kt-outer, shared A fragment, 2 accumulator chains, next-slice preload.
    {
      v8f acc0 = {}, acc1 = {};
      v16bf a  = load_A(xb, 392);
      v16bf b0 = load_B(bp0, 264);
      v16bf b1 = load_B(bp1, 264);
      #pragma unroll
      for (int kt = 0; kt < 8; ++kt) {
        v16bf an = a, b0n = b0, b1n = b1;
        if (kt < 7) {
          an  = load_A(xb + (kt + 1) * 32, 392);
          b0n = load_B(bp0 + (kt + 1) * 32, 264);
          b1n = load_B(bp1 + (kt + 1) * 32, 264);
        }
        acc0 = wmma_bf16(a, b0, acc0);
        acc1 = wmma_bf16(a, b1, acc1);
        a = an; b0 = b0n; b1 = b1n;
      }
      #pragma unroll
      for (int q2 = 0; q2 < 2; ++q2) {
        const v8f& acc = q2 ? acc1 : acc0;
        const int n = (wv * 2 + q2) * 16 + cn;
        if (n < 128) {                     // r gate -> store r*h for c-GEMM
          #pragma unroll
          for (int r = 0; r < 8; ++r) {
            int M = r + mo;
            float rr = sigmoidf(acc[r] + gbs[n]);
            xb[M * 392 + 256 + n] = (__bf16)(rr * hbuf[M * 128 + n]);
          }
        } else {                           // u gate
          int un = n - 128;
          #pragma unroll
          for (int r = 0; r < 8; ++r) {
            int M = r + mo;
            ubuf[M * 128 + un] = sigmoidf(acc[r] + gbs[n]);
          }
        }
      }
    }
    __syncthreads();

    // c = tanh([x, r*h] @ ck + cb); combine. Wave wv owns N-tile wv. Pipelined.
    {
      v8f acc = {};
      v16bf a = load_A(xb, 392);          // kt=0 -> x cols 0..31
      v16bf b = load_B(cp, 264);
      #pragma unroll
      for (int kt = 0; kt < 8; ++kt) {
        v16bf an = a, bn = b;
        if (kt < 7) {
          int col = (kt + 1 < 4) ? (kt + 1) * 32 : (kt + 1) * 32 + 128;  // x region then r*h region
          an = load_A(xb + col, 392);
          bn = load_B(cp + (kt + 1) * 32, 264);
        }
        acc = wmma_bf16(a, b, acc);
        a = an; b = bn;
      }
      const int n = wv * 16 + cn;
      #pragma unroll
      for (int r = 0; r < 8; ++r) {
        int M = r + mo;
        float cc = tanhf(acc[r] + cbs[n]);
        float u = ubuf[M * 128 + n];
        if (ATT) u = (1.f - av[M]) * u;
        float hp = hbuf[M * 128 + n];
        float nh = u * hp + (1.f - u) * cc;
        bool valid = t < sl[M];
        float hn = valid ? nh : hp;
        hbuf[M * 128 + n] = hn;
        xb[M * 392 + 128 + n] = (__bf16)hn;
        if (!ATT) outs[((size_t)(b0 + M) * Tn + t) * 128 + n] = valid ? nh : 0.f;
      }
    }
    __syncthreads();
  }

  if (ATT) {  // final state -> d_out columns 384..511
    for (int i = tid; i < 16 * 128; i += 256) {
      int r = i >> 7, c = i & 127;
      hfin[(size_t)(b0 + r) * 512 + 384 + c] = hbuf[r * 128 + c];
    }
  }
}

// ---------------- attention MLP: one wave per 16-row tile of flattened (b,t) ----------------
__global__ __launch_bounds__(128) void k_attn(
    const float* __restrict__ rnn1, const float* __restrict__ qp,
    const float* __restrict__ w1, const float* __restrict__ b1,
    const float* __restrict__ w2, const float* __restrict__ b2,
    const float* __restrict__ w3, const float* __restrict__ b3,
    const int* __restrict__ mask, float* __restrict__ scores, int Bn, int Tn) {
  extern __shared__ char smem[];
  __bf16* w1T = (__bf16*)smem;              // 80*520  (w1T[n][k] = w1[k][n])
  __bf16* w2T = w1T + 80 * 520;             // 48*104  zero-padded
  __bf16* dinAll = w2T + 48 * 104;          // 4 waves * 16*520
  __bf16* a1All  = dinAll + 4 * 16 * 520;   // 4 waves * 16*104 (zero-padded K)
  float*  a2All  = (float*)(a1All + 4 * 16 * 104); // 4 waves * 16*48
  float*  b1f = a2All + 4 * 16 * 48;        // 80
  float*  b2f = b1f + 80;                   // 40
  float*  w3f = b2f + 40;                   // 40

  const int tid = threadIdx.x, wv = tid >> 5, lane = tid & 31;
  const int cn = lane & 15, mo = (lane >> 4) ? 8 : 0;
  __bf16* din = dinAll + wv * 16 * 520;
  __bf16* a1  = a1All + wv * 16 * 104;
  float*  a2  = a2All + wv * 16 * 48;

  for (int i = tid; i < 80 * 512; i += 128) {
    int n = i >> 9, k = i & 511;
    w1T[n * 520 + k] = (__bf16)w1[k * 80 + n];
  }
  for (int i = tid; i < 48 * 104; i += 128) {
    int n = i / 104, k = i - n * 104;
    float v = (n < 40 && k < 80) ? w2[k * 40 + n] : 0.f;
    w2T[n * 104 + k] = (__bf16)v;
  }
  if (tid < 80) b1f[tid] = b1[tid];
  if (tid < 40) { b2f[tid] = b2[tid]; w3f[tid] = w3[tid]; }
  for (int i = lane; i < 16 * 104; i += 32) a1[i] = (__bf16)0.f;  // zero K-padding

  const long idx0 = ((long)blockIdx.x * 4 + wv) * 16;
  // build din = [qt, rnn1, qt-rnn1, qt*rnn1] for 16 rows
  for (int i = lane; i < 16 * 128; i += 32) {
    int M = i >> 7, c = i & 127;
    long idx = idx0 + M;
    int b = (int)(idx / Tn);
    float qv = qp[(size_t)b * 128 + c];
    float rv = rnn1[idx * 128 + c];
    din[M * 520 + c]       = (__bf16)qv;
    din[M * 520 + 128 + c] = (__bf16)rv;
    din[M * 520 + 256 + c] = (__bf16)(qv - rv);
    din[M * 520 + 384 + c] = (__bf16)(qv * rv);
  }
  __syncthreads();

  // a1 = sigmoid(din @ w1 + b1)   [16x512]@[512x80]
  // kt-outer; ALL 5 B fragments of the K-slice loaded before the WMMA burst so the
  // scheduler can overlap later B-loads with executing WMMAs (partial s_wait_dscnt).
  {
    v8f acc[5];
    #pragma unroll
    for (int nt = 0; nt < 5; ++nt) acc[nt] = (v8f){};
    #pragma unroll
    for (int kt = 0; kt < 16; ++kt) {
      v16bf a = load_A(din + kt * 32, 520);
      v16bf bf0 = load_B(w1T + 0 * 16 * 520 + kt * 32, 520);
      v16bf bf1 = load_B(w1T + 1 * 16 * 520 + kt * 32, 520);
      v16bf bf2 = load_B(w1T + 2 * 16 * 520 + kt * 32, 520);
      v16bf bf3 = load_B(w1T + 3 * 16 * 520 + kt * 32, 520);
      v16bf bf4 = load_B(w1T + 4 * 16 * 520 + kt * 32, 520);
      acc[0] = wmma_bf16(a, bf0, acc[0]);
      acc[1] = wmma_bf16(a, bf1, acc[1]);
      acc[2] = wmma_bf16(a, bf2, acc[2]);
      acc[3] = wmma_bf16(a, bf3, acc[3]);
      acc[4] = wmma_bf16(a, bf4, acc[4]);
    }
    #pragma unroll
    for (int nt = 0; nt < 5; ++nt) {
      int n = nt * 16 + cn;
      #pragma unroll
      for (int r = 0; r < 8; ++r)
        a1[(r + mo) * 104 + n] = (__bf16)sigmoidf(acc[nt][r] + b1f[n]);
    }
  }
  __syncthreads();

  // a2 = sigmoid(a1 @ w2 + b2)   [16x80(pad96)]@[80x40(pad48)] : same grouping, 3 chains
  {
    v8f acc[3];
    #pragma unroll
    for (int nt = 0; nt < 3; ++nt) acc[nt] = (v8f){};
    #pragma unroll
    for (int kt = 0; kt < 3; ++kt) {
      v16bf a = load_A(a1 + kt * 32, 104);
      v16bf bf0 = load_B(w2T + 0 * 16 * 104 + kt * 32, 104);
      v16bf bf1 = load_B(w2T + 1 * 16 * 104 + kt * 32, 104);
      v16bf bf2 = load_B(w2T + 2 * 16 * 104 + kt * 32, 104);
      acc[0] = wmma_bf16(a, bf0, acc[0]);
      acc[1] = wmma_bf16(a, bf1, acc[1]);
      acc[2] = wmma_bf16(a, bf2, acc[2]);
    }
    #pragma unroll
    for (int nt = 0; nt < 3; ++nt) {
      int n = nt * 16 + cn;
      #pragma unroll
      for (int r = 0; r < 8; ++r)
        a2[(r + mo) * 48 + n] = (n < 40) ? sigmoidf(acc[nt][r] + b2f[n]) : 0.f;
    }
  }
  __syncthreads();

  if (lane < 16) {
    int M = lane;
    float s = b3[0];
    for (int n = 0; n < 40; ++n) s += a2[M * 48 + n] * w3f[n];
    long idx = idx0 + M;
    int b = (int)(idx / Tn), t = (int)(idx - (long)b * Tn);
    scores[idx] = (mask[(size_t)b * Tn + t] == 1) ? s : -4294967295.0f;  // -2^32+1
  }
}

// ---------------- softmax over T per batch row ----------------
__global__ __launch_bounds__(256) void k_softmax(const float* __restrict__ scores,
                                                 float* __restrict__ alphas, int Tn) {
  __shared__ float red[256];
  const int b = blockIdx.x, tid = threadIdx.x;
  const float* s = scores + (size_t)b * Tn;
  float m = -3.4e38f;
  for (int t = tid; t < Tn; t += 256) m = fmaxf(m, s[t]);
  red[tid] = m; __syncthreads();
  for (int o = 128; o > 0; o >>= 1) { if (tid < o) red[tid] = fmaxf(red[tid], red[tid + o]); __syncthreads(); }
  m = red[0]; __syncthreads();
  float sum = 0.f;
  for (int t = tid; t < Tn; t += 256) sum += __expf(s[t] - m);
  red[tid] = sum; __syncthreads();
  for (int o = 128; o > 0; o >>= 1) { if (tid < o) red[tid] += red[tid + o]; __syncthreads(); }
  float inv = 1.f / red[0];
  for (int t = tid; t < Tn; t += 256) alphas[(size_t)b * Tn + t] = __expf(s[t] - m) * inv;
}

// ---------------- output cols 0..383: [q | sum | q*sum] ----------------
__global__ void k_assemble(const float* __restrict__ item_eb, const float* __restrict__ his_sum,
                           float* __restrict__ out, int Bn) {
  int i = blockIdx.x * blockDim.x + threadIdx.x;
  if (i >= Bn * 128) return;
  int b = i >> 7, c = i & 127;
  float q = item_eb[i], s = his_sum[i];
  float* o = out + (size_t)b * 512;
  o[c] = q; o[128 + c] = s; o[256 + c] = q * s;
}

extern "C" void kernel_launch(void* const* d_in, const int* in_sizes, int n_in,
                              void* d_out, int out_size, void* d_ws, size_t ws_size,
                              hipStream_t stream) {
  const float* item_eb  = (const float*)d_in[0];
  const float* his      = (const float*)d_in[1];
  const float* his_sum  = (const float*)d_in[2];
  const int*   mask     = (const int*)d_in[3];
  const float* gk1 = (const float*)d_in[4];
  const float* gb1 = (const float*)d_in[5];
  const float* ck1 = (const float*)d_in[6];
  const float* cb1 = (const float*)d_in[7];
  const float* wq  = (const float*)d_in[8];
  const float* bq  = (const float*)d_in[9];
  const float* pal = (const float*)d_in[10];
  const float* w1  = (const float*)d_in[11];
  const float* b1  = (const float*)d_in[12];
  const float* w2  = (const float*)d_in[13];
  const float* b2  = (const float*)d_in[14];
  const float* w3  = (const float*)d_in[15];
  const float* b3  = (const float*)d_in[16];
  const float* gk2 = (const float*)d_in[17];
  const float* gb2 = (const float*)d_in[18];
  const float* ck2 = (const float*)d_in[19];
  const float* cb2 = (const float*)d_in[20];

  const int B = in_sizes[2] / 128;   // item_his_eb_sum = B*128
  const int T = in_sizes[3] / B;     // mask = B*T
  float* out = (float*)d_out;

  char* ws = (char*)d_ws;
  float* rnn1   = (float*)ws; ws += (size_t)B * T * 128 * sizeof(float);
  float* scores = (float*)ws; ws += (size_t)B * T * sizeof(float);
  float* alphas = (float*)ws; ws += (size_t)B * T * sizeof(float);
  float* qp     = (float*)ws; ws += (size_t)B * 128 * sizeof(float);
  int*   seqlen = (int*)ws;   ws += (size_t)B * sizeof(int);

  const size_t gruLds = (size_t)(256 * 264 + 128 * 264 + 16 * 392) * 2 +
                        (size_t)(16 * 128 * 2 + 256 + 128 + 16 + 16) * 4;        // ~233 KB
  const size_t attnLds = (size_t)(80 * 520 + 48 * 104 + 4 * 16 * 520 + 4 * 16 * 104) * 2 +
                         (size_t)(4 * 16 * 48 + 80 + 40 + 40) * 4;               // ~186 KB

  k_seqlen<<<(B + 255) / 256, 256, 0, stream>>>(mask, seqlen, B, T);
  k_qp<<<(B * 128 + 255) / 256, 256, 0, stream>>>(item_eb, wq, bq, pal, qp, B);
  k_gru<false><<<B / 16, 256, gruLds, stream>>>(his, gk1, gb1, ck1, cb1, seqlen,
                                                nullptr, rnn1, nullptr, T);
  k_attn<<<(B * T) / 64, 128, attnLds, stream>>>(rnn1, qp, w1, b1, w2, b2, w3, b3,
                                                 mask, scores, B, T);
  k_softmax<<<B, 256, 0, stream>>>(scores, alphas, T);
  k_gru<true><<<B / 16, 256, gruLds, stream>>>(rnn1, gk2, gb2, ck2, cb2, seqlen,
                                               alphas, nullptr, out, T);
  k_assemble<<<(B * 128 + 255) / 256, 256, 0, stream>>>(item_eb, his_sum, out, B);
}
